// QuantumTransformerWrapper_65481071402939
// MI455X (gfx1250) — compile-verified, matrix operand order checked by reading, not run on hardware
//
#include <hip/hip_runtime.h>
#include <hip/hip_bf16.h>
#include <math.h>

// ---------------------------------------------------------------------------
// Model constants (match reference)
// ---------------------------------------------------------------------------
#define BB   8
#define SS   1024
#define EE   512
#define HH   8          // heads
#define NQ   16
#define FFN  2048
#define NL   4
#define MM   (BB * SS)  // 8192 token rows
#define EPSN 1e-5f

// ---------------------------------------------------------------------------
// CDNA5 WMMA types
// ---------------------------------------------------------------------------
typedef __attribute__((ext_vector_type(16))) __bf16 v16bf;
typedef __attribute__((ext_vector_type(8)))  __bf16 v8bf;
typedef __attribute__((ext_vector_type(8)))  float  v8f;

__device__ __forceinline__ __bf16 f32_to_bf16(float f) {
    union { float f; unsigned u; } v; v.f = f;
    unsigned lsb = (v.u >> 16) & 1u;
    v.u += 0x7FFFu + lsb;                       // round-to-nearest-even
    unsigned short hs = (unsigned short)(v.u >> 16);
    union { unsigned short u; __bf16 b; } r; r.u = hs;
    return r.b;
}

__device__ __forceinline__ v16bf cat16(v8bf lo, v8bf hi) {
    union { v16bf v; v8bf p[2]; } u;
    u.p[0] = lo; u.p[1] = hi;
    return u.v;
}

// ---------------------------------------------------------------------------
// 0) h = x + pe   (pe broadcast over batch)
// ---------------------------------------------------------------------------
__global__ void k_add_pe(const float* __restrict__ x, const float* __restrict__ pe,
                         float* __restrict__ h) {
    int i = blockIdx.x * 256 + threadIdx.x;
    if (i < MM * EE) h[i] = x[i] + pe[i % (SS * EE)];
}

// ---------------------------------------------------------------------------
// 0b) convert all layers' W2 to bf16 once
// ---------------------------------------------------------------------------
__global__ void k_cvt_bf16(const float* __restrict__ in, __bf16* __restrict__ out, int n) {
    int i = blockIdx.x * 256 + threadIdx.x;
    if (i < n) out[i] = f32_to_bf16(in[i]);
}

// ---------------------------------------------------------------------------
// 1) qv[m,j] = cos(theta[j]) * cos(h[m, j])   (j < 16)
// ---------------------------------------------------------------------------
__global__ void k_qfeat(const float* __restrict__ h, const float* __restrict__ theta,
                        float* __restrict__ qv) {
    int i = blockIdx.x * 256 + threadIdx.x;      // over MM*16
    if (i < MM * NQ) {
        int m = i >> 4, j = i & 15;
        qv[i] = __cosf(theta[j]) * __cosf(h[(size_t)m * EE + j]);
    }
}

// ---------------------------------------------------------------------------
// 2) attention, dk == 2, q == k == v.  One block = (batch,head, 256-query chunk).
//    Keys (2 floats each) staged in LDS; online softmax per query.
// ---------------------------------------------------------------------------
__global__ void k_attn(const float* __restrict__ qv, float* __restrict__ attn) {
    __shared__ float kx[SS];
    __shared__ float ky[SS];
    int bh = blockIdx.x;
    int b = bh >> 3, head = bh & 7;
    int c0 = head * 2;
    const float* base = qv + (size_t)b * SS * NQ;
    for (int s = threadIdx.x; s < SS; s += 256) {
        kx[s] = base[s * NQ + c0];
        ky[s] = base[s * NQ + c0 + 1];
    }
    __syncthreads();

    int qi = blockIdx.y * 256 + threadIdx.x;
    float qx = kx[qi], qy = ky[qi];
    const float scale = 0.70710678118654752f;    // 1/sqrt(dk), dk = 2
    float mx = -INFINITY, ssum = 0.f, ax = 0.f, ay = 0.f;
    for (int j = 0; j < SS; ++j) {
        float sc = (qx * kx[j] + qy * ky[j]) * scale;
        float mn = fmaxf(mx, sc);
        float corr = __expf(mx - mn);
        float e    = __expf(sc - mn);
        ssum = ssum * corr + e;
        ax   = ax   * corr + e * kx[j];
        ay   = ay   * corr + e * ky[j];
        mx = mn;
    }
    float inv = 1.0f / ssum;
    size_t o = ((size_t)b * SS + qi) * NQ + c0;
    attn[o]     = ax * inv;
    attn[o + 1] = ay * inv;
}

// ---------------------------------------------------------------------------
// 3) y = h + attn @ Wc^T ; h = LN(y)*g + b
//    One block = 8 tokens; Wc staged once in LDS (stride 17 vs bank conflicts);
//    one wave32 per token, LN stats via wave shuffles (no block barriers).
// ---------------------------------------------------------------------------
__global__ void k_combine_ln(const float* __restrict__ h, const float* __restrict__ attn,
                             const float* __restrict__ Wc,  // [E][16]
                             const float* __restrict__ g, const float* __restrict__ bt,
                             float* __restrict__ hout) {
    __shared__ float wls[EE * 17];               // padded [512][17] -> 34.8 KB
    __shared__ float als[8 * NQ];
    int tid = threadIdx.x;
    int m0 = blockIdx.x * 8;

    for (int i = tid; i < EE * NQ; i += 256) {
        int e = i >> 4, j = i & 15;
        wls[e * 17 + j] = Wc[i];
    }
    if (tid < 8 * NQ) als[tid] = attn[(size_t)m0 * NQ + tid];
    __syncthreads();

    int wave = tid >> 5, lane = tid & 31;
    int m = m0 + wave;
    const float* a = &als[wave * NQ];

    float y[16];
    float lsum = 0.f, lsq = 0.f;
#pragma unroll
    for (int i = 0; i < 16; ++i) {
        int e = lane + i * 32;
        const float* w = &wls[e * 17];
        float acc = h[(size_t)m * EE + e];
#pragma unroll
        for (int j = 0; j < NQ; ++j) acc += a[j] * w[j];
        y[i] = acc;
        lsum += acc; lsq += acc * acc;
    }
#pragma unroll
    for (int off = 16; off > 0; off >>= 1) {
        lsum += __shfl_xor(lsum, off, 32);
        lsq  += __shfl_xor(lsq,  off, 32);
    }
    float mean = lsum * (1.0f / EE);
    float var  = lsq  * (1.0f / EE) - mean * mean;
    float rstd = rsqrtf(var + EPSN);
#pragma unroll
    for (int i = 0; i < 16; ++i) {
        int e = lane + i * 32;
        hout[(size_t)m * EE + e] = (y[i] - mean) * rstd * g[e] + bt[e];
    }
}

// ---------------------------------------------------------------------------
// 4) fused FFN:  t = relu( (cosθ·cos(h[:,:16])) @ W1^T )  -> LDS (bf16)
//               out = t @ W2^T (bf16 WMMA, f32 acc)  ;  h = LN(h + out)
//    One block = 32 token rows, full N = 512, K = 2048.
//    8 waves; wave w owns cols [w*64, w*64+64): per K-step 2 A-frags (LDS),
//    4 B-frags (global), 8 WMMAs -> each B-frag feeds 2 WMMAs (M reuse).
// ---------------------------------------------------------------------------
#define MROWS 32
#define TA_STRIDE 2056   // 2048 + 8 halfs padding (16B) -> rotate LDS banks

__global__ void k_ffn(const float* __restrict__ h, const float* __restrict__ theta,
                      const float* __restrict__ W1,       // [FFN][16] f32
                      const __bf16* __restrict__ W2bf,    // [E][FFN] bf16
                      const float* __restrict__ g, const float* __restrict__ bt,
                      float* __restrict__ hout) {
    __shared__ __align__(16) __bf16 tA[MROWS * TA_STRIDE]; // 131.6 KB, reused as out-tile
    __shared__ float qf[MROWS * NQ];                       // 2 KB
    __shared__ float rs[256], rq[256];
    __shared__ float mrow[MROWS], rrow[MROWS];

    int tid = threadIdx.x;
    int m0 = blockIdx.x * MROWS;

    // --- quantum features for the 32 rows (reads 16 cols of h directly) ---
    for (int i = tid; i < MROWS * NQ; i += 256) {
        int r = i >> 4, j = i & 15;
        qf[i] = __cosf(theta[j]) * __cosf(h[(size_t)(m0 + r) * EE + j]);
    }
    __syncthreads();

    // --- t = relu(qf @ W1^T) into LDS as bf16 -----------------------------
    for (int i = 0; i < 256; ++i) {
        int idx = tid + i * 256;                 // 32*2048 elements
        int r = idx >> 11, k = idx & 2047;
        const float* w = W1 + (size_t)k * NQ;
        const float* q = qf + r * NQ;
        float acc = 0.f;
#pragma unroll
        for (int j = 0; j < NQ; ++j) acc += q[j] * w[j];
        tA[r * TA_STRIDE + k] = f32_to_bf16(fmaxf(acc, 0.f));
    }
    __syncthreads();

    // --- WMMA GEMM: out[32x512] = t[32x2048] @ W2^T -----------------------
    int wave = tid >> 5;
    int lane = tid & 31;
    int kg = lane >> 4;                          // K half-group select
    int ln = lane & 15;                          // row (A) / col (B) in tile
    int nbase = wave * 64;

    v8f zero = {0.f, 0.f, 0.f, 0.f, 0.f, 0.f, 0.f, 0.f};
    v8f acc[2][4];
#pragma unroll
    for (int mt = 0; mt < 2; ++mt)
#pragma unroll
        for (int t = 0; t < 4; ++t) acc[mt][t] = zero;

    const __bf16* aP0 = &tA[(size_t)ln * TA_STRIDE + kg * 8];
    const __bf16* aP1 = &tA[(size_t)(16 + ln) * TA_STRIDE + kg * 8];
    for (int k0 = 0; k0 < FFN; k0 += 32) {
        v16bf a0 = cat16(*(const v8bf*)(aP0 + k0), *(const v8bf*)(aP0 + k0 + 16));
        v16bf a1 = cat16(*(const v8bf*)(aP1 + k0), *(const v8bf*)(aP1 + k0 + 16));
#pragma unroll
        for (int t = 0; t < 4; ++t) {
            const __bf16* bp = W2bf + (size_t)(nbase + t * 16 + ln) * FFN + k0 + kg * 8;
            v16bf bfrag = cat16(*(const v8bf*)bp, *(const v8bf*)(bp + 16));
            acc[0][t] = __builtin_amdgcn_wmma_f32_16x16x32_bf16(
                false, a0, false, bfrag, (short)0, acc[0][t], false, false);
            acc[1][t] = __builtin_amdgcn_wmma_f32_16x16x32_bf16(
                false, a1, false, bfrag, (short)0, acc[1][t], false, false);
        }
    }
    __syncthreads();                             // all waves done reading tA

    // --- residual add into out-tile (reuse tA storage) --------------------
    float* otile = (float*)tA;                   // 32*512 f32 = 64 KB <= 131.6 KB
#pragma unroll
    for (int mt = 0; mt < 2; ++mt)
#pragma unroll
        for (int t = 0; t < 4; ++t) {
            int n = nbase + t * 16 + ln;
#pragma unroll
            for (int r = 0; r < 8; ++r) {
                int row = mt * 16 + r + kg * 8;  // C/D layout: lanes 16-31 hold M+8
                otile[row * EE + n] =
                    h[(size_t)(m0 + row) * EE + n] + acc[mt][t][r];
            }
        }
    __syncthreads();

    // --- LayerNorm over each of the 32 rows (8 threads per row) -----------
    int row = tid >> 3, c0 = tid & 7;
    float lsum = 0.f, lsq = 0.f;
    for (int i = 0; i < 64; ++i) {
        float v = otile[row * EE + c0 + i * 8];
        lsum += v; lsq += v * v;
    }
    rs[tid] = lsum; rq[tid] = lsq;
    __syncthreads();
    if (c0 == 0) {
        float s = 0.f, q = 0.f;
#pragma unroll
        for (int i = 0; i < 8; ++i) { s += rs[row * 8 + i]; q += rq[row * 8 + i]; }
        float mean = s * (1.0f / EE);
        float var  = q * (1.0f / EE) - mean * mean;
        mrow[row] = mean;
        rrow[row] = rsqrtf(var + EPSN);
    }
    __syncthreads();
    float mean = mrow[row], rstd = rrow[row];
    for (int i = 0; i < 64; ++i) {
        int col = c0 + i * 8;
        hout[(size_t)(m0 + row) * EE + col] =
            (otile[row * EE + col] - mean) * rstd * g[col] + bt[col];
    }
}

// ---------------------------------------------------------------------------
// Host launcher
// ---------------------------------------------------------------------------
extern "C" void kernel_launch(void* const* d_in, const int* in_sizes, int n_in,
                              void* d_out, int out_size, void* d_ws, size_t ws_size,
                              hipStream_t stream) {
    (void)in_sizes; (void)n_in; (void)out_size; (void)ws_size;

    const float* x   = (const float*)d_in[0];
    const float* pe  = (const float*)d_in[1];
    const float* tha = (const float*)d_in[2];   // [L,16]
    const float* thf = (const float*)d_in[3];   // [L,16]
    const float* Wc  = (const float*)d_in[4];   // [L,512,16]
    const float* W1  = (const float*)d_in[5];   // [L,2048,16]
    const float* W2  = (const float*)d_in[6];   // [L,512,2048]
    const float* g1  = (const float*)d_in[7];
    const float* b1  = (const float*)d_in[8];
    const float* g2  = (const float*)d_in[9];
    const float* b2  = (const float*)d_in[10];
    float* out = (float*)d_out;

    char* ws = (char*)d_ws;
    float* h    = (float*)ws;  ws += (size_t)MM * EE * sizeof(float);     // 16.8 MB
    float* qv   = (float*)ws;  ws += (size_t)MM * NQ * sizeof(float);     // 0.5 MB
    float* attn = (float*)ws;  ws += (size_t)MM * NQ * sizeof(float);     // 0.5 MB
    __bf16* w2bf = (__bf16*)ws;                                           // 8.4 MB

    k_add_pe<<<(MM * EE) / 256, 256, 0, stream>>>(x, pe, h);
    k_cvt_bf16<<<(NL * EE * FFN) / 256, 256, 0, stream>>>(W2, w2bf, NL * EE * FFN);

    for (int l = 0; l < NL; ++l) {
        k_qfeat<<<(MM * NQ) / 256, 256, 0, stream>>>(h, tha + l * NQ, qv);
        k_attn<<<dim3(BB * HH, SS / 256), 256, 0, stream>>>(qv, attn);
        k_combine_ln<<<MM / 8, 256, 0, stream>>>(h, attn, Wc + (size_t)l * EE * NQ,
                                                 g1 + l * EE, b1 + l * EE, h);
        k_ffn<<<MM / MROWS, 256, 0, stream>>>(h, thf + l * NQ,
                                              W1 + (size_t)l * FFN * NQ,
                                              w2bf + (size_t)l * EE * FFN,
                                              g2 + l * EE, b2 + l * EE,
                                              (l == NL - 1) ? out : h);
    }
}
